// TransformerSequenceLearning_28733331210393
// MI455X (gfx1250) — matrix operand order repaired and checked by reading, not compiled, not run on hardware
//
#include <hip/hip_runtime.h>
#include <hip/hip_bf16.h>
#include <math.h>

typedef __attribute__((ext_vector_type(16))) __bf16 v16bf;
typedef __attribute__((ext_vector_type(8)))  float  v8f;
typedef unsigned short u16;

// CDNA5 async global->LDS staging (guarded: falls back to manual staging)
#if defined(__HIP_DEVICE_COMPILE__) && __has_builtin(__builtin_amdgcn_global_load_async_to_lds_b128)
#define HAVE_ASYNC 1
typedef int vi4 __attribute__((ext_vector_type(4)));
typedef __attribute__((address_space(1))) vi4 g_vi4;
typedef __attribute__((address_space(3))) vi4 l_vi4;
static __device__ __forceinline__ void async_copy16(const void* gsrc, void* ldst) {
  __builtin_amdgcn_global_load_async_to_lds_b128(
      (g_vi4*)(unsigned long long)gsrc,
      (l_vi4*)(unsigned int)(unsigned long long)ldst, 0, 0);
}
static __device__ __forceinline__ void async_wait0() {
#if __has_builtin(__builtin_amdgcn_s_wait_asynccnt)
  __builtin_amdgcn_s_wait_asynccnt(0);
#else
  asm volatile("s_wait_asynccnt 0x0" ::: "memory");
#endif
}
#else
#define HAVE_ASYNC 0
static __device__ __forceinline__ void async_copy16(const void*, void*) {}
static __device__ __forceinline__ void async_wait0() {}
#endif

// problem dims
constexpr int B_ = 8, T_ = 1024, DM_ = 1024, H_ = 16, DH_ = 64, DFF_ = 4096;
constexpr int V_ = 2000, M_ = 256, L_ = 2, DIN_ = 256;

static __device__ __forceinline__ u16 f2bf(float f) {
  unsigned u = __float_as_uint(f);
  unsigned r = u + 0x7FFFu + ((u >> 16) & 1u);
  return (u16)(r >> 16);
}
static __device__ __forceinline__ float bf2f(u16 u) {
  return __uint_as_float(((unsigned)u) << 16);
}

struct U8x16 { uint4 a, b; };   // 32B == v16bf

// ---------------------------------------------------------------------------
// bf16 WMMA GEMM, all operands bf16 in global memory.
//   tile 128x128x32, 256 thr (8 waves), wave 64x32 (8 acc), double-buffered LDS
//   TRANSA: A stored K-major (elem(m,k)=A[k*lda+m])
//   BNM:    B is N-major (B[n*ldb+k]) -> vectorized / async staging
//   epi:    0 none, 1 exact GELU, 2 += resid(f32), 3 *= rowscale[z*sScale+row]
//   outputs: Cf (f32) and/or Cbf (bf16), either may be null
// ---------------------------------------------------------------------------
constexpr int BM = 128, BN = 128, BK = 32;
constexpr int LDSA  = 40;   // ushort stride (80B: conflict-free, 16B-aligned runs)
constexpr int LDSBT = 40;

template <bool TRANSA, bool BNM>
__global__ __launch_bounds__(256)
void gemm_bf16(const u16* __restrict__ A, const u16* __restrict__ Bmat,
               const float* __restrict__ bias, const float* __restrict__ resid,
               const float* __restrict__ rowscale,
               float* __restrict__ Cf, u16* __restrict__ Cbf,
               int Mtot, int Ntot, int K,
               int lda, int ldb, int ldc,
               long long sAo, long long sAi, long long sBo, long long sBi,
               long long sCo, long long sCi, long long sScale,
               int nInner, int epi)
{
  __shared__ alignas(16) u16 sA [2][BM * LDSA];
  __shared__ alignas(16) u16 sBt[2][BN * LDSBT];

  int z  = blockIdx.z;
  int zo = z / nInner, zi = z % nInner;
  const u16* Ag = A    + (long long)zo * sAo + (long long)zi * sAi;
  const u16* Bg = Bmat + (long long)zo * sBo + (long long)zi * sBi;
  long long cOff = (long long)zo * sCo + (long long)zi * sCi;
  float* CfG = Cf  ? Cf  + cOff : nullptr;
  u16*   CbG = Cbf ? Cbf + cOff : nullptr;
  const float* Rg = resid ? resid + cOff : nullptr;

  int mbase = blockIdx.y * BM;
  int nbase = blockIdx.x * BN;

  int tid  = threadIdx.x;
  int lane = tid & 31;
  int wid  = tid >> 5;
  int wrow = wid >> 2;
  int wcol = wid & 3;
  int half = lane >> 4;
  int l16  = lane & 15;

  constexpr bool ASY_A = (!TRANSA) && (HAVE_ASYNC != 0);
  constexpr bool ASY_B = BNM && (HAVE_ASYNC != 0);

  uint4 aR[2], bR[2];

  // zero-fill out-of-bounds chunks once (loop-invariant; never overwritten)
  auto zfill = [&]() {
    #pragma unroll
    for (int i = 0; i < 2; i++) {
      int c = tid + i * 256;
      if constexpr (ASY_A) {
        int r = c >> 2, koff = (c & 3) * 8;
        if (mbase + r >= Mtot) {
          *(uint4*)&sA[0][r * LDSA + koff] = (uint4){0u,0u,0u,0u};
          *(uint4*)&sA[1][r * LDSA + koff] = (uint4){0u,0u,0u,0u};
        }
      }
      if constexpr (ASY_B) {
        int col = c >> 2, koff = (c & 3) * 8;
        if (nbase + col >= Ntot) {
          *(uint4*)&sBt[0][col * LDSBT + koff] = (uint4){0u,0u,0u,0u};
          *(uint4*)&sBt[1][col * LDSBT + koff] = (uint4){0u,0u,0u,0u};
        }
      }
    }
  };

  auto issueA = [&](int kb, int buf) {
    #pragma unroll
    for (int i = 0; i < 2; i++) {
      int c = tid + i * 256;
      int r = c >> 2, koff = (c & 3) * 8;
      int grow = mbase + r;
      if (grow < Mtot)
        async_copy16(Ag + (long long)grow * lda + kb + koff,
                     &sA[buf][r * LDSA + koff]);
    }
  };
  auto issueB = [&](int kb, int buf) {
    #pragma unroll
    for (int i = 0; i < 2; i++) {
      int c = tid + i * 256;
      int col = c >> 2, koff = (c & 3) * 8;
      if (nbase + col < Ntot)
        async_copy16(Bg + (long long)(nbase + col) * ldb + kb + koff,
                     &sBt[buf][col * LDSBT + koff]);
    }
  };

  auto loadA = [&](int kb) {
    #pragma unroll
    for (int i = 0; i < 2; i++) {
      int c = tid + i * 256;
      uint4 v = {0u,0u,0u,0u};
      if (!TRANSA) {
        int r = c >> 2, koff = (c & 3) * 8;
        int grow = mbase + r;
        if (grow < Mtot)
          v = *(const uint4*)(Ag + (long long)grow * lda + kb + koff);
      } else {
        int kk = c >> 4, roff = (c & 15) * 8;
        if (mbase + roff < Mtot)
          v = *(const uint4*)(Ag + (long long)(kb + kk) * lda + mbase + roff);
      }
      aR[i] = v;
    }
  };
  auto storeA = [&](int buf) {
    #pragma unroll
    for (int i = 0; i < 2; i++) {
      int c = tid + i * 256;
      if (!TRANSA) {
        int r = c >> 2, koff = (c & 3) * 8;
        *(uint4*)&sA[buf][r * LDSA + koff] = aR[i];
      } else {
        int kk = c >> 4, roff = (c & 15) * 8;
        u16 tmp[8]; *(uint4*)tmp = aR[i];
        #pragma unroll
        for (int j = 0; j < 8; j++) sA[buf][(roff + j) * LDSA + kk] = tmp[j];
      }
    }
  };
  auto loadB = [&](int kb) {
    #pragma unroll
    for (int i = 0; i < 2; i++) {
      int c = tid + i * 256;
      uint4 v = {0u,0u,0u,0u};
      if (BNM) {
        int col = c >> 2, koff = (c & 3) * 8;
        if (nbase + col < Ntot)
          v = *(const uint4*)(Bg + (long long)(nbase + col) * ldb + kb + koff);
      } else {
        int kk = c >> 4, noff = (c & 15) * 8;
        v = *(const uint4*)(Bg + (long long)(kb + kk) * ldb + nbase + noff);
      }
      bR[i] = v;
    }
  };
  auto storeB = [&](int buf) {
    #pragma unroll
    for (int i = 0; i < 2; i++) {
      int c = tid + i * 256;
      if (BNM) {
        int col = c >> 2, koff = (c & 3) * 8;
        *(uint4*)&sBt[buf][col * LDSBT + koff] = bR[i];
      } else {
        int kk = c >> 4, noff = (c & 15) * 8;
        u16 tmp[8]; *(uint4*)tmp = bR[i];
        #pragma unroll
        for (int j = 0; j < 8; j++) {
          u16 val = (nbase + noff + j < Ntot) ? tmp[j] : (u16)0;
          sBt[buf][(noff + j) * LDSBT + kk] = val;
        }
      }
    }
  };

  v8f acc[4][2];
  #pragma unroll
  for (int i = 0; i < 4; i++)
    #pragma unroll
    for (int j = 0; j < 2; j++)
      acc[i][j] = (v8f){0.f,0.f,0.f,0.f,0.f,0.f,0.f,0.f};

  int nk = K >> 5;

  // prologue: stage tile 0 into buffer 0
  if constexpr (ASY_A || ASY_B) zfill();
  if constexpr (ASY_A) issueA(0, 0); else { loadA(0); storeA(0); }
  if constexpr (ASY_B) issueB(0, 0); else { loadB(0); storeB(0); }

  for (int t = 0; t < nk; ++t) {
    if constexpr (ASY_A || ASY_B) async_wait0();
    __syncthreads();
    int cur = t & 1;
    int nxt = cur ^ 1;
    bool more = (t + 1) < nk;
    int kbn = (t + 1) << 5;

    if (more) {
      if constexpr (ASY_A) issueA(kbn, nxt); else loadA(kbn);
      if constexpr (ASY_B) issueB(kbn, nxt); else loadB(kbn);
    }

    v16bf af[4];
    #pragma unroll
    for (int mt = 0; mt < 4; mt++) {
      int arow = wrow * 64 + mt * 16 + l16;
      const u16* p = &sA[cur][arow * LDSA + half * 8];
      U8x16 tt; tt.a = *(const uint4*)p; tt.b = *(const uint4*)(p + 16);
      af[mt] = __builtin_bit_cast(v16bf, tt);
    }
    v16bf bfv[2];
    #pragma unroll
    for (int nt = 0; nt < 2; nt++) {
      int bcol = wcol * 32 + nt * 16 + l16;
      const u16* p = &sBt[cur][bcol * LDSBT + half * 8];
      U8x16 tt; tt.a = *(const uint4*)p; tt.b = *(const uint4*)(p + 16);
      bfv[nt] = __builtin_bit_cast(v16bf, tt);
    }
    #pragma unroll
    for (int mt = 0; mt < 4; mt++)
      #pragma unroll
      for (int nt = 0; nt < 2; nt++)
        acc[mt][nt] = __builtin_amdgcn_wmma_f32_16x16x32_bf16(
            false, af[mt], false, bfv[nt], (short)0, acc[mt][nt], false, false);

    if (more) {
      if constexpr (!ASY_A) storeA(nxt);
      if constexpr (!ASY_B) storeB(nxt);
    }
  }

  // epilogue + store
  #pragma unroll
  for (int mt = 0; mt < 4; mt++) {
    #pragma unroll
    for (int nt = 0; nt < 2; nt++) {
      int col = nbase + wcol * 32 + nt * 16 + l16;
      if (col >= Ntot) continue;
      float bv = bias ? bias[col] : 0.f;
      #pragma unroll
      for (int e = 0; e < 8; e++) {
        int row = mbase + wrow * 64 + mt * 16 + half * 8 + e;
        if (row >= Mtot) continue;
        float v = acc[mt][nt][e] + bv;
        if (epi == 1) {
          v = 0.5f * v * (1.f + erff(v * 0.70710678118654752f));
        } else if (epi == 2) {
          v += Rg[(long long)row * ldc + col];
        } else if (epi == 3) {
          v *= rowscale[(long long)z * sScale + row];
        }
        if (CfG) CfG[(long long)row * ldc + col] = v;
        if (CbG) CbG[(long long)row * ldc + col] = f2bf(v);
      }
    }
  }
}

// ---------------------------------------------------------------------------
// Support kernels
// ---------------------------------------------------------------------------
__global__ __launch_bounds__(256)
void ln_kernel(const float* __restrict__ x, const float* __restrict__ g,
               const float* __restrict__ b, u16* __restrict__ y, int dim)
{
  int row = blockIdx.x;
  const float* xr = x + (size_t)row * dim;
  u16* yr = y + (size_t)row * dim;
  int tid = threadIdx.x;
  float s = 0.f, s2 = 0.f;
  for (int i = tid; i < dim; i += 256) { float v = xr[i]; s += v; s2 += v * v; }
  __shared__ float rs[256], rs2[256];
  rs[tid] = s; rs2[tid] = s2; __syncthreads();
  for (int off = 128; off > 0; off >>= 1) {
    if (tid < off) { rs[tid] += rs[tid + off]; rs2[tid] += rs2[tid + off]; }
    __syncthreads();
  }
  float mu  = rs[0] / dim;
  float var = rs2[0] / dim - mu * mu;
  float inv = rsqrtf(var + 1e-5f);
  for (int i = tid; i < dim; i += 256)
    yr[i] = f2bf((xr[i] - mu) * inv * g[i] + b[i]);
}

__global__ void add_pe_kernel(float* __restrict__ h, const float* __restrict__ pe, int total) {
  int idx = blockIdx.x * 256 + threadIdx.x;
  if (idx >= total) return;
  int c = idx % DM_;
  int t = (idx / DM_) % T_;
  h[idx] += pe[t * DM_ + c];
}

__global__ void cvt_kernel(const float* __restrict__ in, u16* __restrict__ out, long long n) {
  long long idx = (long long)blockIdx.x * 256 + threadIdx.x;
  if (idx < n) out[idx] = f2bf(in[idx]);
}

__global__ void cvt_scale_kernel(const float* __restrict__ in, u16* __restrict__ out,
                                 long long n, float s) {
  long long idx = (long long)blockIdx.x * 256 + threadIdx.x;
  if (idx < n) out[idx] = f2bf(in[idx] * s);
}

// out[idx] = in[idx*3 + kt]  (slice one tap of Wconv(64,DM,3) -> N-major (64,DM))
__global__ void cvt_stride3_kernel(const float* __restrict__ in, u16* __restrict__ out,
                                   int n, int kt) {
  int idx = blockIdx.x * 256 + threadIdx.x;
  if (idx < n) out[idx] = f2bf(in[(long long)idx * 3 + kt]);
}

// out[n*K+k] = in[k*N+n]  (row-major KxN -> N-major bf16)
__global__ void transpose_cvt_kernel(const float* __restrict__ in, u16* __restrict__ out,
                                     int K, int N) {
  long long idx = (long long)blockIdx.x * 256 + threadIdx.x;
  if (idx >= (long long)K * N) return;
  int n = (int)(idx % N);
  long long k = idx / N;
  out[(long long)n * K + k] = f2bf(in[idx]);
}

__global__ __launch_bounds__(256)
void rowmax_kernel(const float* __restrict__ x, float* __restrict__ out, int rows) {
  int wid = threadIdx.x >> 5, lane = threadIdx.x & 31;
  int row = blockIdx.x * 8 + wid;
  if (row >= rows) return;
  const float* xr = x + (size_t)row * M_;
  float m = -1e30f;
  for (int j = lane; j < M_; j += 32) m = fmaxf(m, xr[j]);
  for (int o = 16; o; o >>= 1) m = fmaxf(m, __shfl_xor(m, o, 32));
  if (lane == 0) out[row] = m;
}

__global__ __launch_bounds__(256)
void segmax_kernel(const float* __restrict__ x, float* __restrict__ out) {
  int bh = blockIdx.x;
  const float* xr = x + (size_t)bh * T_ * M_;
  float m = -1e30f;
  for (int i = threadIdx.x; i < T_ * M_; i += 256) m = fmaxf(m, xr[i]);
  __shared__ float red[256];
  red[threadIdx.x] = m; __syncthreads();
  for (int off = 128; off; off >>= 1) {
    if (threadIdx.x < off) red[threadIdx.x] = fmaxf(red[threadIdx.x], red[threadIdx.x + off]);
    __syncthreads();
  }
  if (!threadIdx.x) out[bh] = red[0];
}

__global__ __launch_bounds__(256)
void diag_kernel(const u16* __restrict__ qkv, float* __restrict__ diag) {
  int wid = threadIdx.x >> 5, lane = threadIdx.x & 31;
  long long row = (long long)blockIdx.x * 8 + wid;
  if (row >= (long long)B_ * H_ * T_) return;
  int t = (int)(row % T_);
  long long bh = row / T_;
  int h = (int)(bh % H_), b = (int)(bh / H_);
  const u16* src = qkv + ((long long)(b * T_ + t) * DM_ + h * DH_);
  float s = 0.f;
  for (int j = lane; j < DH_; j += 32) { float v = bf2f(src[j]); s += v * v; }
  for (int o = 16; o; o >>= 1) s += __shfl_xor(s, o, 32);
  if (!lane) diag[row] = s * 0.0625f;   // 0.5*dn*dn
}

__global__ void featexp_kernel(const float* __restrict__ x, u16* __restrict__ out,
                               const float* __restrict__ diag, const float* __restrict__ stab,
                               long long total, int stabDiv) {
  long long idx = (long long)blockIdx.x * 256 + threadIdx.x;
  if (idx >= total) return;
  long long r = idx >> 8;  // / M_
  float v = x[idx] - diag[r] - stab[idx / stabDiv];
  out[idx] = f2bf(0.0625f * (expf(v) + 1e-4f));   // ratio = M^-0.5
}

__global__ __launch_bounds__(256)
void ksum_kernel(const u16* __restrict__ kp, float* __restrict__ ksum) {
  int bh = blockIdx.x, m = threadIdx.x;
  const u16* src = kp + (size_t)bh * T_ * M_;
  float s = 0.f;
  for (int t = 0; t < T_; t++) s += bf2f(src[t * M_ + m]);
  ksum[bh * M_ + m] = s;
}

__global__ __launch_bounds__(256)
void dinv_kernel(const u16* __restrict__ qp, const float* __restrict__ ksum,
                 float* __restrict__ dinv) {
  int wid = threadIdx.x >> 5, lane = threadIdx.x & 31;
  long long row = (long long)blockIdx.x * 8 + wid;
  if (row >= (long long)B_ * H_ * T_) return;
  long long bh = row / T_;
  const u16* q   = qp + row * M_;
  const float* ks = ksum + bh * M_;
  float s = 0.f;
  for (int j = lane; j < M_; j += 32) s += bf2f(q[j]) * ks[j];
  for (int o = 16; o; o >>= 1) s += __shfl_xor(s, o, 32);
  if (!lane) dinv[row] = 1.f / s;
}

// exact recompute of conv outputs at sequence edges t==0 and t==T-1
__global__ void conv_fix_kernel(const float* __restrict__ h, const float* __restrict__ W,
                                const float* __restrict__ b, u16* __restrict__ out) {
  int idx = blockIdx.x * 256 + threadIdx.x;
  if (idx >= B_ * 2 * 64) return;
  int oc = idx & 63;
  int e  = (idx >> 6) & 1;
  int bb = idx >> 7;
  int t  = e ? (T_ - 1) : 0;
  const float* w = W + (size_t)oc * DM_ * 3;
  float acc = b[oc];
  for (int kt = 0; kt < 3; kt++) {
    int tt = t + kt - 1;
    if (tt < 0 || tt >= T_) continue;
    const float* hr = h + (long long)(bb * T_ + tt) * DM_;
    for (int ic = 0; ic < DM_; ic++) acc += hr[ic] * w[ic * 3 + kt];
  }
  out[((long long)bb * T_ + t) * 64 + oc] = f2bf(acc);
}

// ---------------------------------------------------------------------------
// Host side
// ---------------------------------------------------------------------------
static inline void launch_gemm(hipStream_t s, const u16* A, const u16* Bm,
    const float* bias, const float* resid, const float* rowscale,
    float* Cf, u16* Cbf,
    int M, int N, int K, int lda, int ldb, int ldc,
    long long sAo, long long sAi, long long sBo, long long sBi,
    long long sCo, long long sCi, long long sScale,
    int nInner, int nOuter, int flags, int epi)
{
  dim3 grid((N + BN - 1) / BN, (M + BM - 1) / BM, nOuter * nInner);
  if (flags == 2)
    gemm_bf16<false, true><<<grid, dim3(256), 0, s>>>(A, Bm, bias, resid, rowscale, Cf, Cbf,
        M, N, K, lda, ldb, ldc, sAo, sAi, sBo, sBi, sCo, sCi, sScale, nInner, epi);
  else if (flags == 1)
    gemm_bf16<true, false><<<grid, dim3(256), 0, s>>>(A, Bm, bias, resid, rowscale, Cf, Cbf,
        M, N, K, lda, ldb, ldc, sAo, sAi, sBo, sBi, sCo, sCi, sScale, nInner, epi);
  else
    gemm_bf16<false, false><<<grid, dim3(256), 0, s>>>(A, Bm, bias, resid, rowscale, Cf, Cbf,
        M, N, K, lda, ldb, ldc, sAo, sAi, sBo, sBi, sCo, sCi, sScale, nInner, epi);
}

// workspace layout (bytes)
constexpr size_t PAD     = 4096;
constexpr size_t F32_ACT = (size_t)B_ * T_ * DM_ * 4;
constexpr size_t F32_PHI = (size_t)B_ * H_ * T_ * (size_t)M_ * 4;
constexpr size_t BF_ACT  = (size_t)B_ * T_ * DM_ * 2;
constexpr size_t BF_PHI  = (size_t)B_ * H_ * T_ * (size_t)M_ * 2;

constexpr size_t O_H     = 0;
constexpr size_t O_DDQ   = O_H + F32_ACT + PAD;
constexpr size_t O_DDK   = O_DDQ + F32_PHI + PAD;
constexpr size_t O_KSUM  = O_DDK + F32_PHI + PAD;
constexpr size_t O_STABQ = O_KSUM + (size_t)B_ * H_ * M_ * 4 + PAD;
constexpr size_t O_STABK = O_STABQ + (size_t)B_ * H_ * T_ * 4 + PAD;
constexpr size_t O_DIAGQ = O_STABK + 4096 + PAD;
constexpr size_t O_DIAGK = O_DIAGQ + (size_t)B_ * H_ * T_ * 4 + PAD;
constexpr size_t O_DINV  = O_DIAGK + (size_t)B_ * H_ * T_ * 4 + PAD;
constexpr size_t O_XBF   = O_DINV + (size_t)B_ * H_ * T_ * 4 + PAD;
constexpr size_t O_WINT  = O_XBF + (size_t)B_ * T_ * DIN_ * 2 + PAD;
constexpr size_t O_WQT   = O_WINT + (size_t)DIN_ * DM_ * 2 + PAD;
constexpr size_t O_WKT   = O_WQT + (size_t)L_ * DM_ * DM_ * 2 + PAD;
constexpr size_t O_WVT   = O_WKT + (size_t)L_ * DM_ * DM_ * 2 + PAD;
constexpr size_t O_WOT   = O_WVT + (size_t)L_ * DM_ * DM_ * 2 + PAD;
constexpr size_t O_W1T   = O_WOT + (size_t)L_ * DM_ * DM_ * 2 + PAD;
constexpr size_t O_W2T   = O_W1T + (size_t)L_ * DM_ * DFF_ * 2 + PAD;
constexpr size_t O_WCT   = O_W2T + (size_t)L_ * DFF_ * DM_ * 2 + PAD;
constexpr size_t O_WAUXT = O_WCT + (size_t)V_ * DM_ * 2 + PAD;
constexpr size_t O_WCV3  = O_WAUXT + (size_t)(V_ + 1) * 64 * 2 + PAD;
constexpr size_t O_PRB   = O_WCV3 + (size_t)3 * 64 * DM_ * 2 + PAD;
constexpr size_t O_YBF   = O_PRB + (size_t)M_ * DH_ * 2 + PAD;
constexpr size_t O_BIGBF = O_YBF + BF_ACT + PAD;
constexpr size_t O_QPBF  = O_BIGBF + (size_t)B_ * T_ * DFF_ * 2 + PAD;
constexpr size_t O_KPBF  = O_QPBF + BF_PHI + PAD;
constexpr size_t O_CTXBF = O_KPBF + BF_PHI + PAD;
constexpr size_t O_HBF   = O_CTXBF + (size_t)B_ * H_ * M_ * DH_ * 2 + PAD;
constexpr size_t O_CONVF = O_HBF + BF_ACT + PAD;
constexpr size_t O_CONVBF= O_CONVF + (size_t)B_ * T_ * 64 * 4 + PAD;

extern "C" void kernel_launch(void* const* d_in, const int* in_sizes, int n_in,
                              void* d_out, int out_size, void* d_ws, size_t ws_size,
                              hipStream_t stream)
{
  (void)in_sizes; (void)n_in; (void)out_size; (void)ws_size;
  const float* x    = (const float*)d_in[0];
  const float* Win  = (const float*)d_in[1];
  const float* b_in = (const float*)d_in[2];
  const float* pe   = (const float*)d_in[3];
  const float* ln1g = (const float*)d_in[4];
  const float* ln1b = (const float*)d_in[5];
  const float* Wq   = (const float*)d_in[6];
  const float* bq   = (const float*)d_in[7];
  const float* Wk   = (const float*)d_in[8];
  const float* bk   = (const float*)d_in[9];
  const float* Wv   = (const float*)d_in[10];
  const float* bv   = (const float*)d_in[11];
  const float* Wo   = (const float*)d_in[12];
  const float* bo   = (const float*)d_in[13];
  const float* ln2g = (const float*)d_in[14];
  const float* ln2b = (const float*)d_in[15];
  const float* W1   = (const float*)d_in[16];
  const float* b1   = (const float*)d_in[17];
  const float* W2   = (const float*)d_in[18];
  const float* b2   = (const float*)d_in[19];
  const float* Wc   = (const float*)d_in[20];
  const float* bc   = (const float*)d_in[21];
  const float* Wcv  = (const float*)d_in[22];
  const float* bcv  = (const float*)d_in[23];
  const float* Waux = (const float*)d_in[24];
  const float* baux = (const float*)d_in[25];
  const float* proj = (const float*)d_in[26];

  char* ws = (char*)d_ws;
  float* h     = (float*)(ws + O_H);
  float* ddq   = (float*)(ws + O_DDQ);
  float* ddk   = (float*)(ws + O_DDK);
  float* ksum  = (float*)(ws + O_KSUM);
  float* stabq = (float*)(ws + O_STABQ);
  float* stabk = (float*)(ws + O_STABK);
  float* diagq = (float*)(ws + O_DIAGQ);
  float* diagk = (float*)(ws + O_DIAGK);
  float* dinv  = (float*)(ws + O_DINV);
  u16* xbf   = (u16*)(ws + O_XBF);
  u16* WinT  = (u16*)(ws + O_WINT);
  u16* WqT   = (u16*)(ws + O_WQT);
  u16* WkT   = (u16*)(ws + O_WKT);
  u16* WvT   = (u16*)(ws + O_WVT);
  u16* WoT   = (u16*)(ws + O_WOT);
  u16* W1T   = (u16*)(ws + O_W1T);
  u16* W2T   = (u16*)(ws + O_W2T);
  u16* WcT   = (u16*)(ws + O_WCT);
  u16* WauxT = (u16*)(ws + O_WAUXT);
  u16* Wcv3  = (u16*)(ws + O_WCV3);
  u16* prB   = (u16*)(ws + O_PRB);
  u16* ybf   = (u16*)(ws + O_YBF);
  u16* qbf   = (u16*)(ws + O_BIGBF);
  u16* kbf   = (u16*)(ws + O_BIGBF) + BF_ACT / 2;
  u16* vbf   = (u16*)(ws + O_BIGBF) + BF_ACT;
  u16* oabf  = (u16*)(ws + O_BIGBF) + BF_ACT / 2 * 3;
  u16* ffbf  = (u16*)(ws + O_BIGBF);
  u16* qpbf  = (u16*)(ws + O_QPBF);
  u16* kpbf  = (u16*)(ws + O_KPBF);
  u16* ctxbf = (u16*)(ws + O_CTXBF);
  u16* hbf   = (u16*)(ws + O_HBF);
  float* convf = (float*)(ws + O_CONVF);
  u16* cvbf  = (u16*)(ws + O_CONVBF);

  float* gloss = (float*)d_out;
  float* aux   = gloss + (size_t)B_ * T_ * V_;

  const int rowsBT = B_ * T_;
  const long long zTM = (long long)T_ * M_;
  const long long zMD = (long long)M_ * DH_;

  // ---- one-time conversions (bf16, weights pre-transposed to N-major) ----
  auto cvt = [&](const float* in, u16* out, long long n) {
    cvt_kernel<<<(unsigned)((n + 255) / 256), 256, 0, stream>>>(in, out, n);
  };
  auto tcvt = [&](const float* in, u16* out, int K, int N) {
    long long n = (long long)K * N;
    transpose_cvt_kernel<<<(unsigned)((n + 255) / 256), 256, 0, stream>>>(in, out, K, N);
  };
  cvt(x, xbf, (long long)rowsBT * DIN_);
  tcvt(Win, WinT, DIN_, DM_);
  for (int l = 0; l < L_; l++) {
    size_t wo = (size_t)l * DM_ * DM_;
    tcvt(Wq + wo, WqT + wo, DM_, DM_);
    tcvt(Wk + wo, WkT + wo, DM_, DM_);
    tcvt(Wv + wo, WvT + wo, DM_, DM_);
    tcvt(Wo + wo, WoT + wo, DM_, DM_);
    tcvt(W1 + (size_t)l * DM_ * DFF_, W1T + (size_t)l * DM_ * DFF_, DM_, DFF_);
    tcvt(W2 + (size_t)l * DFF_ * DM_, W2T + (size_t)l * DFF_ * DM_, DFF_, DM_);
  }
  tcvt(Wc, WcT, DM_, V_);
  tcvt(Waux, WauxT, 64, V_ + 1);
  for (int kt = 0; kt < 3; kt++)
    cvt_stride3_kernel<<<(64 * DM_ + 255) / 256, 256, 0, stream>>>(
        Wcv, Wcv3 + (size_t)kt * 64 * DM_, 64 * DM_, kt);
  cvt_scale_kernel<<<(M_ * DH_ + 255) / 256, 256, 0, stream>>>(
      proj, prB, (long long)M_ * DH_, 0.35355339059327373f);   // dn = DH^-0.25

  // ---- embedding ----
  launch_gemm(stream, xbf, WinT, b_in, nullptr, nullptr, h, nullptr,
              rowsBT, DM_, DIN_, DIN_, DIN_, DM_, 0,0,0,0,0,0,0, 1,1, 2, 0);
  add_pe_kernel<<<(rowsBT * DM_) / 256, 256, 0, stream>>>(h, pe, rowsBT * DM_);

  for (int l = 0; l < L_; l++) {
    size_t wo = (size_t)l * DM_ * DM_;
    ln_kernel<<<rowsBT, 256, 0, stream>>>(h, ln1g + l * DM_, ln1b + l * DM_, ybf, DM_);
    launch_gemm(stream, ybf, WqT + wo, bq + l * DM_, nullptr, nullptr, nullptr, qbf,
                rowsBT, DM_, DM_, DM_, DM_, DM_, 0,0,0,0,0,0,0, 1,1, 2, 0);
    launch_gemm(stream, ybf, WkT + wo, bk + l * DM_, nullptr, nullptr, nullptr, kbf,
                rowsBT, DM_, DM_, DM_, DM_, DM_, 0,0,0,0,0,0,0, 1,1, 2, 0);
    launch_gemm(stream, ybf, WvT + wo, bv + l * DM_, nullptr, nullptr, nullptr, vbf,
                rowsBT, DM_, DM_, DM_, DM_, DM_, 0,0,0,0,0,0,0, 1,1, 2, 0);

    // dd = q|k @ (dn * proj^T) : batched over (b,h), B = prB is N-major (= proj itself)
    launch_gemm(stream, qbf, prB, nullptr, nullptr, nullptr, ddq, nullptr,
                T_, M_, DH_, DM_, DH_, M_,
                (long long)T_ * DM_, DH_, 0, 0,
                (long long)H_ * zTM, zTM, 0, H_, B_, 2, 0);
    launch_gemm(stream, kbf, prB, nullptr, nullptr, nullptr, ddk, nullptr,
                T_, M_, DH_, DM_, DH_, M_,
                (long long)T_ * DM_, DH_, 0, 0,
                (long long)H_ * zTM, zTM, 0, H_, B_, 2, 0);

    diag_kernel<<<B_ * H_ * T_ / 8, 256, 0, stream>>>(qbf, diagq);
    diag_kernel<<<B_ * H_ * T_ / 8, 256, 0, stream>>>(kbf, diagk);
    rowmax_kernel<<<B_ * H_ * T_ / 8, 256, 0, stream>>>(ddq, stabq, B_ * H_ * T_);
    segmax_kernel<<<B_ * H_, 256, 0, stream>>>(ddk, stabk);

    long long totPhi = (long long)B_ * H_ * T_ * M_;
    featexp_kernel<<<(unsigned)(totPhi / 256), 256, 0, stream>>>(ddq, qpbf, diagq, stabq, totPhi, M_);
    featexp_kernel<<<(unsigned)(totPhi / 256), 256, 0, stream>>>(ddk, kpbf, diagk, stabk, totPhi, T_ * M_);
    ksum_kernel<<<B_ * H_, 256, 0, stream>>>(kpbf, ksum);

    // ctx = kp^T @ v  (transA; B row-major activation)
    launch_gemm(stream, kpbf, vbf, nullptr, nullptr, nullptr, nullptr, ctxbf,
                M_, DH_, T_, M_, DM_, DH_,
                (long long)H_ * zTM, zTM,
                (long long)T_ * DM_, DH_,
                (long long)H_ * zMD, zMD, 0, H_, B_, 1, 0);

    dinv_kernel<<<B_ * H_ * T_ / 8, 256, 0, stream>>>(qpbf, ksum, dinv);

    // o = (qp @ ctx) * dinv[row] -> (B,T,H,DH)
    launch_gemm(stream, qpbf, ctxbf, nullptr, nullptr, dinv, nullptr, oabf,
                T_, DH_, M_, M_, DH_, DM_,
                (long long)H_ * zTM, zTM,
                (long long)H_ * zMD, zMD,
                (long long)T_ * DM_, DH_, T_, H_, B_, 0, 3);

    // h += o @ Wo + bo
    launch_gemm(stream, oabf, WoT + wo, bo + l * DM_, h, nullptr, h, nullptr,
                rowsBT, DM_, DM_, DM_, DM_, DM_, 0,0,0,0,0,0,0, 1,1, 2, 2);

    // FFN
    ln_kernel<<<rowsBT, 256, 0, stream>>>(h, ln2g + l * DM_, ln2b + l * DM_, ybf, DM_);
    launch_gemm(stream, ybf, W1T + (size_t)l * DM_ * DFF_, b1 + l * DFF_, nullptr, nullptr,
                nullptr, ffbf,
                rowsBT, DFF_, DM_, DM_, DM_, DFF_, 0,0,0,0,0,0,0, 1,1, 2, 1);
    launch_gemm(stream, ffbf, W2T + (size_t)l * DFF_ * DM_, b2 + l * DM_, h, nullptr,
                h, hbf,
                rowsBT, DM_, DFF_, DFF_, DFF_, DM_, 0,0,0,0,0,0,0, 1,1, 2, 2);

    if (l == 0) {
      // conv as 3 shifted WMMA GEMMs accumulating through the resid epilogue
      launch_gemm(stream, hbf - DM_, Wcv3, bcv, nullptr, nullptr, convf, nullptr,
                  rowsBT, 64, DM_, DM_, DM_, 64, 0,0,0,0,0,0,0, 1,1, 2, 0);
      launch_gemm(stream, hbf, Wcv3 + (size_t)64 * DM_, nullptr, convf, nullptr, convf, nullptr,
                  rowsBT, 64, DM_, DM_, DM_, 64, 0,0,0,0,0,0,0, 1,1, 2, 2);
      launch_gemm(stream, hbf + DM_, Wcv3 + (size_t)2 * 64 * DM_, nullptr, convf, nullptr,
                  nullptr, cvbf,
                  rowsBT, 64, DM_, DM_, DM_, 64, 0,0,0,0,0,0,0, 1,1, 2, 2);
      conv_fix_kernel<<<(B_ * 2 * 64 + 255) / 256, 256, 0, stream>>>(h, Wcv, bcv, cvbf);
      launch_gemm(stream, cvbf, WauxT, baux, nullptr, nullptr, aux, nullptr,
                  rowsBT, V_ + 1, 64, 64, 64, V_ + 1, 0,0,0,0,0,0,0, 1,1, 2, 0);
    }
  }

  // gloss = h @ Wc + bc
  launch_gemm(stream, hbf, WcT, bc, nullptr, nullptr, gloss, nullptr,
              rowsBT, V_, DM_, DM_, DM_, V_, 0,0,0,0,0,0,0, 1,1, 2, 0);
}
